// MSFNO1d_78357383348738
// MI455X (gfx1250) — compile-verified, matrix operand order checked by reading, not run on hardware
//
#include <hip/hip_runtime.h>
#include <hip/hip_bf16.h>

typedef _Float16 h16;
typedef __attribute__((ext_vector_type(16))) _Float16 v16h;
typedef __attribute__((ext_vector_type(8)))  float    v8f;
typedef __attribute__((ext_vector_type(4)))  float    f32x4;

#define HCH   64
#define BSZ   32
#define LLEN  4096
#define LOG2L 12
#define NMOD  2048
#define KG    8
#define IC    32

__device__ __forceinline__ float fast_gelu(float x) {
    float x3 = x * x * x;
    return 0.5f * x * (1.0f + tanhf(0.7978845608028654f * (x + 0.044715f * x3)));
}

__device__ __forceinline__ v8f vzero() {
    v8f z;
#pragma unroll
    for (int q = 0; q < 8; ++q) z[q] = 0.f;
    return z;
}

// Fragment swizzle for 16-bit WMMA operands (K within a 32-wide k-step):
// low lanes hold K in [0,8)u[16,24); high lanes K in [8,16)u[24,32).
__device__ __forceinline__ int fr32(int i) {
    return (((i >> 3) & 1) << 4) | (i & 7) | (((i >> 4) & 1) << 3);
}
__device__ __forceinline__ int fr64(int i) {
    return (i >> 5) * 32 + fr32(i & 31);
}

// ---------------- encoder: v[b,h,l] = enc_w[h,:] . [x;u](b,:,l) + enc_b[h] ----
__global__ void k_encoder(const float* __restrict__ x, const float* __restrict__ u,
                          const float* __restrict__ ew, const float* __restrict__ eb,
                          float* __restrict__ v) {
    int l = blockIdx.x * 256 + threadIdx.x;
    int b = blockIdx.y;
    float xv = x[(size_t)b * LLEN + l];
    float u0 = u[((size_t)b * 3 + 0) * LLEN + l];
    float u1 = u[((size_t)b * 3 + 1) * LLEN + l];
    float u2 = u[((size_t)b * 3 + 2) * LLEN + l];
    for (int h = 0; h < HCH; ++h) {
        float val = ew[h * 4 + 0] * xv + ew[h * 4 + 1] * u0 +
                    ew[h * 4 + 2] * u1 + ew[h * 4 + 3] * u2 + eb[h];
        v[((size_t)b * HCH + h) * LLEN + l] = val;
    }
}

// ---------------- forward real FFT (length 4096, keep bins 0..2047) ----------
__global__ void k_fft_fwd(const float* __restrict__ vin, float2* __restrict__ X) {
    __shared__ __align__(16) float2 s[LLEN];
    __shared__ __align__(16) float2 tw[LLEN / 2];   // e^{-2*pi*i*k/4096}
    int row = blockIdx.x;
    const float* src = vin + (size_t)row * LLEN;
    int tid = threadIdx.x;
    for (int t = 0; t < 8; ++t) {
        int k = tid + t * 256;
        float ang = -3.14159265358979323846f * (float)k / 2048.0f;
        float sn, cs; __sincosf(ang, &sn, &cs);
        tw[k] = make_float2(cs, sn);
    }
    for (int t = 0; t < 16; ++t) {
        int idx = tid + t * 256;
        int rev = __brev((unsigned)idx) >> (32 - LOG2L);
        s[rev] = make_float2(src[idx], 0.0f);
    }
    __syncthreads();
    for (int st = 0; st < LOG2L; ++st) {
        int half = 1 << st;
        int shift = (LOG2L - 1) - st;
        for (int t = 0; t < 8; ++t) {
            int bf = tid + t * 256;
            int j  = bf & (half - 1);
            int i0 = ((bf >> st) << (st + 1)) + j;
            int i1 = i0 + half;
            float2 wv = tw[j << shift];
            float2 a = s[i0], bb = s[i1];
            float2 tt = make_float2(wv.x * bb.x - wv.y * bb.y,
                                    wv.x * bb.y + wv.y * bb.x);
            s[i0] = make_float2(a.x + tt.x, a.y + tt.y);
            s[i1] = make_float2(a.x - tt.x, a.y - tt.y);
        }
        __syncthreads();
    }
    float2* dst = X + (size_t)row * NMOD;
    for (int t = 0; t < 8; ++t) { int k = tid + t * 256; dst[k] = s[k]; }
}

// ---------------- inverse FFT (Hermitian extend 2048 bins -> 4096 real) ------
__global__ void k_fft_inv(const float2* __restrict__ Y, float* __restrict__ S,
                          const float* __restrict__ sb0, const float* __restrict__ sb1,
                          const float* __restrict__ sb2) {
    __shared__ __align__(16) float2 s[LLEN];
    __shared__ __align__(16) float2 tw[LLEN / 2];   // forward table; conjugated at use
    int row = blockIdx.x;
    int o = row & (HCH - 1);
    float bias = sb0[o] + sb1[o] + sb2[o];
    const float2* src = Y + (size_t)row * NMOD;
    int tid = threadIdx.x;
    for (int t = 0; t < 8; ++t) {
        int k = tid + t * 256;
        float ang = -3.14159265358979323846f * (float)k / 2048.0f;
        float sn, cs; __sincosf(ang, &sn, &cs);
        tw[k] = make_float2(cs, sn);
    }
    for (int t = 0; t < 16; ++t) {
        int idx = tid + t * 256;
        float2 val;
        if (idx < NMOD)        val = src[idx];
        else if (idx == NMOD)  val = make_float2(0.f, 0.f);
        else { float2 c = src[LLEN - idx]; val = make_float2(c.x, -c.y); }
        int rev = __brev((unsigned)idx) >> (32 - LOG2L);
        s[rev] = val;
    }
    __syncthreads();
    for (int st = 0; st < LOG2L; ++st) {
        int half = 1 << st;
        int shift = (LOG2L - 1) - st;
        for (int t = 0; t < 8; ++t) {
            int bf = tid + t * 256;
            int j  = bf & (half - 1);
            int i0 = ((bf >> st) << (st + 1)) + j;
            int i1 = i0 + half;
            float2 wv = tw[j << shift];             // use conj(wv) = (x, -y)
            float2 a = s[i0], bb = s[i1];
            float2 tt = make_float2(wv.x * bb.x + wv.y * bb.y,
                                    wv.x * bb.y - wv.y * bb.x);
            s[i0] = make_float2(a.x + tt.x, a.y + tt.y);
            s[i1] = make_float2(a.x - tt.x, a.y - tt.y);
        }
        __syncthreads();
    }
    float* dst = S + (size_t)row * LLEN;
    const float inv = 1.0f / (float)LLEN;
    for (int t = 0; t < 16; ++t) {
        int idx = tid + t * 256;
        dst[idx] = s[idx].x * inv + bias;
    }
}

// ---------------- spectral complex GEMM: Y[b,o,k] = sum_i X[b,i,k]*Weff[i,o,k]
// block: 8 waves <-> 8 consecutive modes; per mode M=32(b), N=64(o), K=64(i)
__global__ void __launch_bounds__(256) k_spec(const float2* __restrict__ X,
        const float* __restrict__ w0, const float* __restrict__ w1,
        const float* __restrict__ w2, float2* __restrict__ Y) {
    __shared__ __align__(64) unsigned char smem[57344];
    h16*    Xl = (h16*)smem;                 // [kk][2 part][32 b][32 swz(i)]  32KB
    h16*    Wl = (h16*)(smem + 32768);       // [kk][3 part][16 o][32 swz(i)]  24KB
    float2* Ol = (float2*)smem;              // reuse: [kk][32 b][16 o]        32KB

    int tid  = threadIdx.x;
    int lane = tid & 31;
    int kk   = tid >> 5;              // wave id == mode offset in group
    int k0   = blockIdx.x * KG;

    v8f cre[2][4], cim[2][4];
#pragma unroll
    for (int m = 0; m < 2; ++m)
#pragma unroll
        for (int n = 0; n < 4; ++n) { cre[m][n] = vzero(); cim[m][n] = vzero(); }

    int hi = lane >> 4;
    int lo = lane & 15;

    for (int ic = 0; ic < 2; ++ic) {
        int i0 = ic * IC;
        // ---- stage X chunk: (b:32, i':32, k:8) ; 64B contiguous per (b,i) ----
        for (int t = 0; t < 4; ++t) {
            int p  = tid + t * 256;
            int b  = p >> 5, ii = p & 31;
            const float4* xp = (const float4*)(X + ((size_t)b * HCH + (i0 + ii)) * NMOD + k0);
            float re[8], im[8];
#pragma unroll
            for (int j = 0; j < 4; ++j) {
                float4 q = xp[j];
                re[2 * j] = q.x; im[2 * j] = q.y; re[2 * j + 1] = q.z; im[2 * j + 1] = q.w;
            }
            int fo = fr32(ii);
#pragma unroll
            for (int m = 0; m < 8; ++m) {
                Xl[((m * 2 + 0) * BSZ + b) * 32 + fo] = (h16)re[m];
                Xl[((m * 2 + 1) * BSZ + b) * 32 + fo] = (h16)im[m];
            }
        }
#pragma unroll
        for (int nt = 0; nt < 4; ++nt) {
            int o0 = nt * 16;
            // ---- stage Weff chunk: sum applicable scales, f32->f16 ----
            // Weights are pure streaming (no reuse, > L2 capacity): use NT loads.
            for (int t = 0; t < 2; ++t) {
                int p  = tid + t * 256;
                int oo = p & 15, ii = p >> 4;
                float re[8], im[8];
                {
                    const f32x4* wp = (const f32x4*)(w2 +
                        (((size_t)(i0 + ii) * HCH + (o0 + oo)) * 2048 + k0) * 2);
#pragma unroll
                    for (int j = 0; j < 4; ++j) {
                        f32x4 q = __builtin_nontemporal_load(wp + j);
                        re[2 * j] = q.x; im[2 * j] = q.y; re[2 * j + 1] = q.z; im[2 * j + 1] = q.w;
                    }
                }
                if (k0 < 1024) {
                    const f32x4* wp = (const f32x4*)(w1 +
                        (((size_t)(i0 + ii) * HCH + (o0 + oo)) * 1024 + k0) * 2);
#pragma unroll
                    for (int j = 0; j < 4; ++j) {
                        f32x4 q = __builtin_nontemporal_load(wp + j);
                        re[2 * j] += q.x; im[2 * j] += q.y; re[2 * j + 1] += q.z; im[2 * j + 1] += q.w;
                    }
                }
                if (k0 < 512) {
                    const f32x4* wp = (const f32x4*)(w0 +
                        (((size_t)(i0 + ii) * HCH + (o0 + oo)) * 512 + k0) * 2);
#pragma unroll
                    for (int j = 0; j < 4; ++j) {
                        f32x4 q = __builtin_nontemporal_load(wp + j);
                        re[2 * j] += q.x; im[2 * j] += q.y; re[2 * j + 1] += q.z; im[2 * j + 1] += q.w;
                    }
                }
                // prefetch the next chunk of the dominant stream (w2)
                {
                    int c = ic * 4 + nt;
                    if (c < 7) {
                        int cn  = c + 1;
                        int icn = cn >> 2, ntn = cn & 3;
                        const float* np = w2 +
                            (((size_t)(icn * IC + ii) * HCH + (ntn * 16 + oo)) * 2048 + k0) * 2;
                        __builtin_prefetch(np, 0, 1);
                    }
                }
                int fo = fr32(ii);
#pragma unroll
                for (int m = 0; m < 8; ++m) {
                    Wl[((m * 3 + 0) * 16 + oo) * 32 + fo] = (h16)re[m];
                    Wl[((m * 3 + 1) * 16 + oo) * 32 + fo] = (h16)im[m];
                    Wl[((m * 3 + 2) * 16 + oo) * 32 + fo] = (h16)(-im[m]);
                }
            }
            __syncthreads();
            // ---- WMMA: Yre = Xre*Wre + Xim*(-Wim); Yim = Xre*Wim + Xim*Wre ----
            v16h bre = *(const v16h*)&Wl[((kk * 3 + 0) * 16 + lo) * 32 + hi * 16];
            v16h bim = *(const v16h*)&Wl[((kk * 3 + 1) * 16 + lo) * 32 + hi * 16];
            v16h bni = *(const v16h*)&Wl[((kk * 3 + 2) * 16 + lo) * 32 + hi * 16];
#pragma unroll
            for (int m = 0; m < 2; ++m) {
                int b = m * 16 + lo;
                v16h are = *(const v16h*)&Xl[((kk * 2 + 0) * BSZ + b) * 32 + hi * 16];
                v16h aim = *(const v16h*)&Xl[((kk * 2 + 1) * BSZ + b) * 32 + hi * 16];
                cre[m][nt] = __builtin_amdgcn_wmma_f32_16x16x32_f16(false, are, false, bre,
                                 (short)0, cre[m][nt], false, false);
                cre[m][nt] = __builtin_amdgcn_wmma_f32_16x16x32_f16(false, aim, false, bni,
                                 (short)0, cre[m][nt], false, false);
                cim[m][nt] = __builtin_amdgcn_wmma_f32_16x16x32_f16(false, are, false, bim,
                                 (short)0, cim[m][nt], false, false);
                cim[m][nt] = __builtin_amdgcn_wmma_f32_16x16x32_f16(false, aim, false, bre,
                                 (short)0, cim[m][nt], false, false);
            }
            __syncthreads();
        }
    }
    // ---- transpose outputs through LDS; 64B-contiguous Y stores over k ----
#pragma unroll
    for (int nt = 0; nt < 4; ++nt) {
#pragma unroll
        for (int m = 0; m < 2; ++m) {
            int bb = m * 16 + hi * 8;
#pragma unroll
            for (int g = 0; g < 8; ++g)
                Ol[((size_t)kk * BSZ + bb + g) * 16 + lo] =
                    make_float2(cre[m][nt][g], cim[m][nt][g]);
        }
        __syncthreads();
        for (int t = 0; t < 2; ++t) {
            int p = tid + t * 256;
            int b = p >> 4, oo = p & 15;
            float2* dst = Y + ((size_t)b * HCH + (nt * 16 + oo)) * NMOD + k0;
#pragma unroll
            for (int m8 = 0; m8 < 8; ++m8)
                dst[m8] = Ol[((size_t)m8 * BSZ + b) * 16 + oo];
        }
        __syncthreads();
    }
}

// ---------------- fused conv1 -> gelu -> conv2 -> residual (WMMA) ------------
__global__ void __launch_bounds__(256) k_mlp(const float* __restrict__ S,
        const float* __restrict__ w1, const float* __restrict__ b1,
        const float* __restrict__ w2, const float* __restrict__ b2,
        float* __restrict__ v) {
    __shared__ __align__(64) h16 Sl[128 * 80];
    __shared__ __align__(64) h16 Hl[128 * 80];
    __shared__ __align__(64) h16 W1l[64 * 80];
    __shared__ __align__(64) h16 W2l[64 * 80];
    int tid = threadIdx.x, lane = tid & 31, w = tid >> 5;
    int b  = blockIdx.y;
    int l0 = blockIdx.x * 128;
    for (int t = 0; t < 16; ++t) {
        int e = tid + t * 256;
        int o = e >> 6, i = e & 63;
        W1l[o * 80 + fr64(i)] = (h16)w1[e];
        W2l[o * 80 + fr64(i)] = (h16)w2[e];
    }
    for (int t = 0; t < 32; ++t) {
        int e = tid + t * 256;
        int i = e >> 7, lp = e & 127;
        Sl[lp * 80 + fr64(i)] = (h16)S[((size_t)b * HCH + i) * LLEN + l0 + lp];
    }
    __syncthreads();
    int mt = w & 3;
    int nb = (w >> 2) * 4;
    int hi = lane >> 4, lo = lane & 15;
    // GEMM1: h = gelu(W1 . s + b1)
    for (int nt = nb; nt < nb + 4; ++nt) {
        v8f acc = vzero();
#pragma unroll
        for (int ks = 0; ks < 2; ++ks) {
            int kb = ks * 32 + hi * 16;
            v16h A  = *(const v16h*)&W1l[(mt * 16 + lo) * 80 + kb];
            v16h Bf = *(const v16h*)&Sl[(nt * 16 + lo) * 80 + kb];
            acc = __builtin_amdgcn_wmma_f32_16x16x32_f16(false, A, false, Bf,
                      (short)0, acc, false, false);
        }
        int lp = nt * 16 + lo;
#pragma unroll
        for (int g = 0; g < 8; ++g) {
            int o = mt * 16 + g + 8 * hi;
            Hl[lp * 80 + fr64(o)] = (h16)fast_gelu(acc[g] + b1[o]);
        }
    }
    __syncthreads();
    // GEMM2: v += W2 . h + b2
    for (int nt = nb; nt < nb + 4; ++nt) {
        v8f acc = vzero();
#pragma unroll
        for (int ks = 0; ks < 2; ++ks) {
            int kb = ks * 32 + hi * 16;
            v16h A  = *(const v16h*)&W2l[(mt * 16 + lo) * 80 + kb];
            v16h Bf = *(const v16h*)&Hl[(nt * 16 + lo) * 80 + kb];
            acc = __builtin_amdgcn_wmma_f32_16x16x32_f16(false, A, false, Bf,
                      (short)0, acc, false, false);
        }
#pragma unroll
        for (int g = 0; g < 8; ++g) {
            int o = mt * 16 + g + 8 * hi;
            size_t idx = ((size_t)b * HCH + o) * LLEN + l0 + nt * 16 + lo;
            v[idx] = v[idx] + acc[g] + b2[o];
        }
    }
}

// ---------------- decoder ----------------------------------------------------
__global__ void k_decoder(const float* __restrict__ v, const float* __restrict__ dw,
                          const float* __restrict__ db, float* __restrict__ out) {
    int l = blockIdx.x * 256 + threadIdx.x;
    int b = blockIdx.y;
    float acc = db[0];
    for (int h = 0; h < HCH; ++h)
        acc += dw[h] * v[((size_t)b * HCH + h) * LLEN + l];
    out[(size_t)b * LLEN + l] = acc;
}

extern "C" void kernel_launch(void* const* d_in, const int* in_sizes, int n_in,
                              void* d_out, int out_size, void* d_ws, size_t ws_size,
                              hipStream_t stream) {
    (void)in_sizes; (void)n_in; (void)out_size; (void)ws_size;
    const float* u     = (const float*)d_in[0];
    const float* x     = (const float*)d_in[1];
    const float* enc_w = (const float*)d_in[2];
    const float* enc_b = (const float*)d_in[3];
    const float* dec_w = (const float*)d_in[4];
    const float* dec_b = (const float*)d_in[5];
    const float* c1w   = (const float*)d_in[6];
    const float* c1b   = (const float*)d_in[7];
    const float* c2w   = (const float*)d_in[8];
    const float* c2b   = (const float*)d_in[9];
    const float* sw0   = (const float*)d_in[10];
    const float* sb0   = (const float*)d_in[11];
    const float* sw1   = (const float*)d_in[12];
    const float* sb1   = (const float*)d_in[13];
    const float* sw2   = (const float*)d_in[14];
    const float* sb2   = (const float*)d_in[15];
    float* out = (float*)d_out;

    char* ws = (char*)d_ws;
    const size_t MB32 = (size_t)32 * 1024 * 1024;
    float*  v = (float*)ws;                    // (B,H,L) f32
    float2* X = (float2*)(ws + MB32);          // (B,H,2048) c64  (aliased by S)
    float*  S = (float*)(ws + MB32);           // (B,H,L) f32     (X dead by then)
    float2* Y = (float2*)(ws + 2 * MB32);      // (B,H,2048) c64

    k_encoder<<<dim3(16, 32), 256, 0, stream>>>(x, u, enc_w, enc_b, v);
    for (int l = 0; l < 4; ++l) {
        k_fft_fwd<<<2048, 256, 0, stream>>>(v, X);
        k_spec<<<256, 256, 0, stream>>>(X,
            sw0 + (size_t)l * HCH * HCH * 512 * 2,
            sw1 + (size_t)l * HCH * HCH * 1024 * 2,
            sw2 + (size_t)l * HCH * HCH * 2048 * 2, Y);
        k_fft_inv<<<2048, 256, 0, stream>>>(Y, S, sb0 + l * HCH, sb1 + l * HCH, sb2 + l * HCH);
        k_mlp<<<dim3(32, 32), 256, 0, stream>>>(S, c1w + (size_t)l * HCH * HCH, c1b + l * HCH,
                                                c2w + (size_t)l * HCH * HCH, c2b + l * HCH, v);
    }
    k_decoder<<<dim3(16, 32), 256, 0, stream>>>(v, dec_w, dec_b, out);
}
